// T5MultiHeadAttention_1932735283622
// MI455X (gfx1250) — matrix operand order repaired and checked by reading, not compiled, not run on hardware
//
#include <hip/hip_runtime.h>
#include <hip/hip_bf16.h>

typedef __attribute__((ext_vector_type(16))) _Float16 v16h;
typedef __attribute__((ext_vector_type(8)))  _Float16 h8;
typedef __attribute__((ext_vector_type(8)))  float    v8f;
typedef __attribute__((ext_vector_type(2)))  _Float16 h2;
typedef __attribute__((ext_vector_type(2)))  float    f2;

#define B_    2
#define S_    2048
#define H_    16
#define C_    64
#define WAVES 8
#define JT    64   // keys per iteration
#define KP    72   // row stride in halves: 144B = 16B-aligned, 36 banks (conflict-free)
#define VP    72
#define PP    72

#define CAT16(lo, hi) __builtin_shufflevector(lo, hi, 0,1,2,3,4,5,6,7,8,9,10,11,12,13,14,15)

// Flash-attention for T5 (no 1/sqrt(c) scale, additive bias), f16 WMMA / f32 accum.
__global__ __launch_bounds__(256) void t5_mha_fa_kernel(
    const float* __restrict__ q, const float* __restrict__ k,
    const float* __restrict__ v, const float* __restrict__ bias,
    float* __restrict__ out)
{
    __shared__ __align__(16) _Float16 lK[JT * KP];          // K tile, [j_local][c]
    __shared__ __align__(16) _Float16 lV[64 * VP];          // V^T tile, [c][j_local]
    __shared__ __align__(16) _Float16 lP[WAVES][16 * PP];   // per-wave P, [i_local][j_local]

    const int tid  = threadIdx.x;
    const int lane = tid & 31;
    const int w    = tid >> 5;
    const int g    = lane >> 4;    // lane half: rows 0-7 vs 8-15 in C/D layout
    const int n    = lane & 15;    // column index within 16 (C/D & B), row for A
    const int h    = blockIdx.y;
    const int b    = blockIdx.z;
    const int i0   = (blockIdx.x * WAVES + w) * 16;

    // ---- Q once as two f16 A-operands (c=0..31, 32..63) ----------------------
    // A 16x32 f16: element e <- K=16g+e (contiguous runs)
    v16h aq[2];
    {
        const float* qr = q + ((size_t)(b * S_ + (i0 + n)) * H_ + h) * C_;
#pragma unroll
        for (int ch = 0; ch < 2; ++ch) {
#pragma unroll
            for (int half = 0; half < 2; ++half) {
                const float* src = qr + 32 * ch + 8 * half + 16 * g;
#pragma unroll
                for (int e = 0; e < 8; ++e)
                    aq[ch][8 * half + e] = (_Float16)src[e];   // 8 contiguous f32 -> b128s
            }
        }
    }

    // Constant all-ones B operand: accsum = P x ones gives row sums via WMMA.
    v16h bones;
#pragma unroll
    for (int e = 0; e < 16; ++e) bones[e] = (_Float16)1.0f;

    // ---- State: O accumulators (16x64), row-sum accumulator, running max -----
    v8f acc[4], accsum;
    float m_run = -__builtin_inff();
#pragma unroll
    for (int r = 0; r < 8; ++r) {
        acc[0][r] = 0.0f; acc[1][r] = 0.0f; acc[2][r] = 0.0f; acc[3][r] = 0.0f;
        accsum[r] = 0.0f;
    }

    // ---- Loop-carried streaming pointers (all in-loop offsets are immediates)
    const int jj0 = tid >> 5, c2s = tid & 31;     // K staging coords (const per thread)
    const int jp0 = tid >> 6, ccs = tid & 63;     // V staging coords (const per thread)
    const float* kptr = k + ((size_t)(b * S_ + jj0) * H_ + h) * C_ + 2 * c2s;
    const float* vptr = v + ((size_t)(b * S_ + 2 * jp0) * H_ + h) * C_ + ccs;
    const float* bptr = bias + ((size_t)h * S_ + (size_t)(i0 + 8 * g)) * S_ + n;
    _Float16* lkst = &lK[jj0 * KP + 2 * c2s];
    _Float16* lvst = &lV[ccs * VP + 2 * jp0];

    // ---- Stream over keys, 64 at a time --------------------------------------
    for (int it = 0; it < S_ / JT; ++it) {
        __syncthreads();   // previous iteration's K/V reads done before overwrite

        // Stage K (64x64) as f16: float2 loads, packed h2 stores (imm offsets).
#pragma unroll
        for (int r = 0; r < 8; ++r) {
            const f2 kf = *(const f2*)(kptr + r * 8 * H_ * C_);   // rows jj0+8r
            h2 hk; hk[0] = (_Float16)kf[0]; hk[1] = (_Float16)kf[1];
            *(h2*)(lkst + r * 8 * KP) = hk;
        }
        // Stage V transposed (64x64): two strided b32 loads -> one packed store.
#pragma unroll
        for (int r = 0; r < 8; ++r) {
            h2 hv;
            hv[0] = (_Float16)vptr[r * 8 * H_ * C_];              // row 2*(jp0+4r)
            hv[1] = (_Float16)vptr[r * 8 * H_ * C_ + H_ * C_];    // row 2*(jp0+4r)+1
            *(h2*)(lvst + r * 8) = hv;
        }
        __syncthreads();

        // ---- S = Q*K^T + bias (four 16x16 tiles) -----------------------------
        v8f s[4];
#pragma unroll
        for (int jt = 0; jt < 4; ++jt) {
            v8f cb;   // bias straight into C-layout (non-temporal: streamed once)
#pragma unroll
            for (int r = 0; r < 8; ++r)
                cb[r] = __builtin_nontemporal_load(bptr + r * S_ + jt * 16);
            // B 32x16: lane col N=lane%16, k = 16g + e (16 contiguous halves -> 2x b128)
            const _Float16* krow = &lK[(jt * 16 + n) * KP + 16 * g];
            const v16h bk0 = CAT16(*(const h8*)(krow),      *(const h8*)(krow + 8));
            const v16h bk1 = CAT16(*(const h8*)(krow + 32), *(const h8*)(krow + 40));
            cb = __builtin_amdgcn_wmma_f32_16x16x32_f16(false, aq[0], false, bk0,
                                                        (short)0, cb, false, false);
            cb = __builtin_amdgcn_wmma_f32_16x16x32_f16(false, aq[1], false, bk1,
                                                        (short)0, cb, false, false);
            s[jt] = cb;
        }

        // ---- Block-uniform online max (softmax is shift-invariant per row) ---
        float mb = fmaxf(fmaxf(s[0][0], s[1][0]), fmaxf(s[2][0], s[3][0]));
#pragma unroll
        for (int r = 1; r < 8; ++r)
            mb = fmaxf(mb, fmaxf(fmaxf(s[0][r], s[1][r]), fmaxf(s[2][r], s[3][r])));
#pragma unroll
        for (int d = 1; d < 32; d <<= 1) mb = fmaxf(mb, __shfl_xor(mb, d, 32));

        const float m_new = fmaxf(m_run, mb);
        if (mb > m_run) {              // wave-uniform branch; rarely taken late
            const float al = __expf(m_run - m_new);   // first iter: exp(-inf)=0
#pragma unroll
            for (int r = 0; r < 8; ++r) {
                acc[0][r] *= al; acc[1][r] *= al;
                acc[2][r] *= al; acc[3][r] *= al;
                accsum[r] *= al;
            }
        }
        m_run = m_new;

        // ---- P = exp(S - m), staged to per-wave LDS tile (C -> A layout) -----
#pragma unroll
        for (int jt = 0; jt < 4; ++jt) {
#pragma unroll
            for (int r = 0; r < 8; ++r) {
                const float p = __expf(s[jt][r] - m_run);
                lP[w][(8 * g + r) * PP + jt * 16 + n] = (_Float16)p;
            }
        }
        // Same-wave DS ops are in-order (ISA ch.11); only block compiler reordering.
        asm volatile("" ::: "memory");

        v16h ap[2];   // P 16x64 as two 16x32 A operands (2x b128 each)
#pragma unroll
        for (int jh = 0; jh < 2; ++jh) {
            const _Float16* prow = &lP[w][n * PP + 16 * g + 32 * jh];
            ap[jh] = CAT16(*(const h8*)(prow), *(const h8*)(prow + 8));
        }

        // ---- O += P*V (4 tiles x 2 halves) and row-sums += P*ones ------------
#pragma unroll
        for (int ct = 0; ct < 4; ++ct) {
#pragma unroll
            for (int jh = 0; jh < 2; ++jh) {
                const _Float16* vrow = &lV[(ct * 16 + n) * VP + 32 * jh + 16 * g];
                const v16h bv = CAT16(*(const h8*)(vrow), *(const h8*)(vrow + 8));
                acc[ct] = __builtin_amdgcn_wmma_f32_16x16x32_f16(false, ap[jh], false, bv,
                                                                 (short)0, acc[ct], false, false);
            }
        }
        accsum = __builtin_amdgcn_wmma_f32_16x16x32_f16(false, ap[0], false, bones,
                                                        (short)0, accsum, false, false);
        accsum = __builtin_amdgcn_wmma_f32_16x16x32_f16(false, ap[1], false, bones,
                                                        (short)0, accsum, false, false);

        // advance streaming pointers (single 64-bit add each)
        kptr += JT * H_ * C_;
        vptr += JT * H_ * C_;
        bptr += JT;
    }

    // ---- Epilogue: normalize by WMMA-computed row sums, NT stores ------------
#pragma unroll
    for (int r = 0; r < 8; ++r) {
        const float inv = 1.0f / accsum[r];
        float* orow = out + ((size_t)(b * S_ + (i0 + 8 * g + r)) * H_ + h) * C_;
#pragma unroll
        for (int ct = 0; ct < 4; ++ct)
            __builtin_nontemporal_store(acc[ct][r] * inv, orow + ct * 16 + n);
    }
}

extern "C" void kernel_launch(void* const* d_in, const int* in_sizes, int n_in,
                              void* d_out, int out_size, void* d_ws, size_t ws_size,
                              hipStream_t stream) {
    (void)in_sizes; (void)n_in; (void)out_size; (void)d_ws; (void)ws_size;
    const float* q    = (const float*)d_in[0];
    const float* k    = (const float*)d_in[1];
    const float* v    = (const float*)d_in[2];
    const float* bias = (const float*)d_in[3];
    float* out = (float*)d_out;

    dim3 grid(S_ / (16 * WAVES), H_, B_);   // 16 x 16 x 2 = 512 workgroups
    dim3 block(256);                        // 8 wave32 waves
    hipLaunchKernelGGL(t5_mha_fa_kernel, grid, block, 0, stream, q, k, v, bias, out);
}
